// monkey_switch_QKV_27066883899582
// MI455X (gfx1250) — matrix operand order repaired and checked by reading, not compiled, not run on hardware
//
#include <hip/hip_runtime.h>
#include <stdint.h>

typedef __bf16 bf16t;
typedef __attribute__((ext_vector_type(16))) __bf16 v16bf;
typedef __attribute__((ext_vector_type(8)))  float  v8f;

constexpr int BATCH = 256;
constexpr int CIN   = 2304;
constexpr int CH    = 512;          // HIDDEN
constexpr int HWXY  = 49;           // 7*7
constexpr int PH    = 9;            // padded spatial
constexpr int M_CONV = BATCH * HWXY;   // 12544
constexpr int K_RED  = CIN * 9;        // 20736
constexpr int K_CH   = CH * 9;         // 4608

union FragU { uint4 u[2]; v16bf v; };

// CDNA5 async copy: global memory -> LDS, no VGPR staging, tracked by ASYNCcnt.
__device__ __forceinline__ void async_copy_b128(const void* gaddr, uint32_t lds_addr) {
  asm volatile("global_load_async_to_lds_b128 %0, %1, off"
               :: "v"(lds_addr), "v"(gaddr) : "memory");
}
__device__ __forceinline__ void wait_async0() {
  asm volatile("s_wait_asynccnt 0x0" ::: "memory");
}

struct GemmP {
  const bf16t* A;
  const bf16t* Bt;       // [N][K], K contiguous
  float* outF;
  bf16t* outB;
  const float* resid;
  int M, N, K;
  long zA, zB;           // per-blockIdx.z element strides
  float scale;
};

// MODE: 0 reduce-conv, 1 k-conv, 2 v-conv, 3 att, 4 virt, 5 out-conv
template<int MODE>
__global__ __launch_bounds__(256)
void gemm_bf16_wmma(GemmP p) {
  constexpr bool IS_CONV = (MODE==0)||(MODE==1)||(MODE==2)||(MODE==5);
  constexpr int  CINM    = (MODE==0) ? CIN : CH;   // conv input channels (compile-time)

  __shared__ __align__(16) bf16t lA[2][128][32];
  __shared__ __align__(16) bf16t lB[2][128][32];

  const int tid  = threadIdx.x;
  const int lane = tid & 31;
  const int wid  = tid >> 5;        // 0..7
  const int wm   = wid >> 1;        // 0..3 : 32-row slice of M
  const int wn   = wid & 1;         // 0..1 : 64-col slice of N
  const int m0   = blockIdx.x * 128;
  const int n0   = blockIdx.y * 128;
  const int z    = blockIdx.z;

  const bf16t* Abase = p.A  + (long)z * p.zA;
  const bf16t* Bbase = p.Bt + (long)z * p.zB;

  const int lrow  = tid >> 1;       // 0..127
  const int lhalf = tid & 1;        // which 16-element k-half

  int ab = 0, aoh = 0, aow = 0;
  const int gma = m0 + lrow;
  if (IS_CONV) {
    ab = gma / 49; int hw = gma - ab*49; aoh = hw / 7; aow = hw - aoh*7;
  }
  const int gnb = n0 + lrow;

  v8f acc[2][4];
  #pragma unroll
  for (int i=0;i<2;++i)
    #pragma unroll
    for (int j=0;j<4;++j)
      #pragma unroll
      for (int e=0;e<8;++e) acc[i][j][e] = 0.f;

  const int nk = p.K >> 5;

  auto issueTiles = [&](int kt, int buf) {
    const int k0 = kt << 5;
    const bf16t* pa;
    if (IS_CONV) {
      const int t9  = k0 / CINM;             // tap index (div by constant)
      const int kh  = t9 / 3, kw = t9 - kh*3;
      const int ci0 = k0 - t9 * CINM;
      pa = Abase + ((long)(ab*9 + aoh + kh) * 9 + (aow + kw)) * CINM + ci0 + lhalf*16;
    } else {
      pa = Abase + (long)gma * p.K + k0 + lhalf*16;
    }
    const bf16t* pb = Bbase + (long)gnb * p.K + k0 + lhalf*16;
    const uint32_t la = (uint32_t)(uintptr_t)&lA[buf][lrow][lhalf*16];
    const uint32_t lb = (uint32_t)(uintptr_t)&lB[buf][lrow][lhalf*16];
    async_copy_b128(pa,     la);
    async_copy_b128(pa + 8, la + 16);
    async_copy_b128(pb,     lb);
    async_copy_b128(pb + 8, lb + 16);
  };

  issueTiles(0, 0);
  wait_async0();
  __syncthreads();

  const int fr    = lane & 15;
  const int hiH   = lane >> 4;      // 0/1 lane half
  const int koffA = hiH * 8;        // A frag: K 0-7 / 8-15 (+16 block)
  const int koffB = hiH * 8;        // B frag: K 0-7 / 8-15 (+16 block)

  for (int kt = 0; kt < nk; ++kt) {
    const int cur = kt & 1;
    const bool more = (kt + 1 < nk);
    if (more) issueTiles(kt + 1, cur ^ 1);

    v16bf af[2], bfr[4];
    #pragma unroll
    for (int i=0;i<2;++i) {
      const int r = wm*32 + i*16 + fr;
      FragU u;
      u.u[0] = *(const uint4*)&lA[cur][r][koffA];
      u.u[1] = *(const uint4*)&lA[cur][r][16 + koffA];
      af[i] = u.v;
    }
    #pragma unroll
    for (int j=0;j<4;++j) {
      const int c = wn*64 + j*16 + fr;
      FragU u;
      u.u[0] = *(const uint4*)&lB[cur][c][koffB];
      u.u[1] = *(const uint4*)&lB[cur][c][16 + koffB];
      bfr[j] = u.v;
    }
    #pragma unroll
    for (int i=0;i<2;++i)
      #pragma unroll
      for (int j=0;j<4;++j)
        acc[i][j] = __builtin_amdgcn_wmma_f32_16x16x32_bf16(
            false, af[i], false, bfr[j], (short)0, acc[i][j], false, false);

    if (more) wait_async0();
    __syncthreads();
  }

  #pragma unroll
  for (int i=0;i<2;++i) {
    #pragma unroll
    for (int j=0;j<4;++j) {
      #pragma unroll
      for (int e=0;e<8;++e) {
        const int gm = m0 + wm*32 + i*16 + e + hiH*8;
        const int gn = n0 + wn*64 + j*16 + fr;
        const float val = acc[i][j][e] * p.scale;
        if (MODE == 0) {
          const int b = gm / 49; const int hw = gm - b*49;
          const int oh = hw / 7, ow = hw - oh*7;
          p.outF[(((long)b*CH + gn)*7 + oh)*7 + ow] = val;                       // x f32 NCHW
          p.outB[(((long)b*9 + oh + 1)*9 + (ow + 1))*CH + gn] = (bf16t)val;      // padded NHWC bf16
        } else if (MODE == 1) {
          const int b = gm / 49; const int hw = gm - b*49;
          p.outB[((long)hw*BATCH + b)*CH + gn] = (bf16t)val;                     // kbuf[hw][j][c]
        } else if (MODE == 2) {
          const int b = gm / 49; const int hw = gm - b*49;
          p.outB[((long)hw*CH + gn)*BATCH + b] = (bf16t)val;                     // vt[hw][c][j]
        } else if (MODE == 3) {
          p.outF[((long)z*BATCH + gm)*BATCH + gn] = val;                         // att[hw][b][j]
        } else if (MODE == 4) {
          p.outF[((long)gm*HWXY + z)*CH + gn] = val;                             // virt[b][hw][c]
        } else {
          const int b = gm / 49; const int hw = gm - b*49;
          const int oh = hw / 7, ow = hw - oh*7;
          const long idx = (((long)b*CH + gn)*7 + oh)*7 + ow;
          p.outF[idx] = p.resid[idx] + val;                                      // x + conv(virt)
        }
      }
    }
  }
}

__global__ void zero_b16(uint4* p, long n16) {
  long i = (long)blockIdx.x*256 + threadIdx.x;
  if (i < n16) p[i] = make_uint4(0,0,0,0);
}

__global__ void fill_xpad1(const float* x, bf16t* xp) {
  long idx = (long)blockIdx.x*256 + threadIdx.x;
  if (idx >= (long)BATCH*CIN*HWXY) return;
  int hw = (int)(idx % 49); long r = idx / 49;
  int ci = (int)(r % CIN);  int b = (int)(r / CIN);
  int h = hw / 7, w = hw - h*7;
  xp[(((long)b*9 + h + 1)*9 + (w + 1))*CIN + ci] = (bf16t)x[idx];
}

__global__ void transpose_w(const float* w, bf16t* wT, int cin) {
  long idx = (long)blockIdx.x*256 + threadIdx.x;
  long total = (long)CH * cin * 9;
  if (idx >= total) return;
  int t9 = (int)(idx % 9); long r = idx / 9;
  int ci = (int)(r % cin); int co = (int)(r / cin);
  wT[(long)co*cin*9 + (long)t9*cin + ci] = (bf16t)w[idx];   // [co][(kh*3+kw)*cin+ci]
}

__global__ void compute_q(const float* status, const float* rois, const float* wq, bf16t* qbuf) {
  long idx = (long)blockIdx.x*256 + threadIdx.x;   // ((b*49)+hw)*512 + c
  if (idx >= (long)BATCH*HWXY*CH) return;
  int c = (int)(idx & 511); long r = idx >> 9;
  int hw = (int)(r % 49); int b = (int)(r / 49);
  int oh = hw / 7, ow = hw - oh*7;
  float s0 = status[b*2 + 0];
  float s1 = rois[b*2 + 1];
  float acc = 0.f;
  for (int kh = 0; kh < 3; ++kh) {
    int ih = oh + kh - 1; if (ih < 0 || ih >= 7) continue;
    for (int kw = 0; kw < 3; ++kw) {
      int iw = ow + kw - 1; if (iw < 0 || iw >= 7) continue;
      acc += s0 * wq[((c*2 + 0)*3 + kh)*3 + kw]
           + s1 * wq[((c*2 + 1)*3 + kh)*3 + kw];
    }
  }
  qbuf[((long)hw*BATCH + b)*CH + c] = (bf16t)acc;           // qbuf[hw][b][c]
}

__global__ void softmax_rows(const float* att, bf16t* attb) {
  __shared__ float red[256];
  const int row = blockIdx.x;            // z*256 + b
  const int t = threadIdx.x;
  float v = att[(long)row*256 + t];
  red[t] = v; __syncthreads();
  for (int s = 128; s > 0; s >>= 1) { if (t < s) red[t] = fmaxf(red[t], red[t+s]); __syncthreads(); }
  float mx = red[0]; __syncthreads();
  float e = expf(v - mx);
  red[t] = e; __syncthreads();
  for (int s = 128; s > 0; s >>= 1) { if (t < s) red[t] += red[t+s]; __syncthreads(); }
  attb[(long)row*256 + t] = (bf16t)(e / red[0]);
}

__global__ void layernorm_relu_pad(const float* virt, const float* gamma, const float* beta,
                                   bf16t* outpad) {
  __shared__ float s1[256], s2[256];
  const int b = blockIdx.x, t = threadIdx.x;
  const long base = (long)b * HWXY * CH;
  float sum = 0.f, sq = 0.f;
  for (int i = t; i < HWXY*CH; i += 256) { float v = virt[base + i]; sum += v; sq += v*v; }
  s1[t] = sum; s2[t] = sq; __syncthreads();
  for (int s = 128; s > 0; s >>= 1) { if (t < s) { s1[t] += s1[t+s]; s2[t] += s2[t+s]; } __syncthreads(); }
  const float n = (float)(HWXY*CH);
  const float mu = s1[0] / n;
  const float var = s2[0] / n - mu*mu;
  const float rinv = rsqrtf(var + 1e-5f);
  __syncthreads();
  for (int i = t; i < HWXY*CH; i += 256) {
    int hw = i >> 9, c = i & 511;
    float v = (virt[base + i] - mu) * rinv * gamma[c] + beta[c];
    v = fmaxf(v, 0.f);
    int oh = hw / 7, ow = hw - oh*7;
    outpad[(((long)b*9 + oh + 1)*9 + (ow + 1))*CH + c] = (bf16t)v;
  }
}

extern "C" void kernel_launch(void* const* d_in, const int* in_sizes, int n_in,
                              void* d_out, int out_size, void* d_ws, size_t ws_size,
                              hipStream_t stream) {
  (void)in_sizes; (void)n_in; (void)out_size; (void)ws_size;
  const float* status = (const float*)d_in[0];
  const float* rois   = (const float*)d_in[1];
  const float* bbox   = (const float*)d_in[2];
  const float* w_red  = (const float*)d_in[3];
  const float* w_q    = (const float*)d_in[4];
  const float* w_k    = (const float*)d_in[5];
  const float* w_v    = (const float*)d_in[6];
  const float* w_out  = (const float*)d_in[7];
  const float* gamma  = (const float*)d_in[8];
  const float* beta   = (const float*)d_in[9];
  float* out = (float*)d_out;

  char* ws = (char*)d_ws;
  size_t off = 0;
  auto alloc = [&](size_t bytes) { void* p = ws + off; off += (bytes + 255) & ~(size_t)255; return p; };

  bf16t* xpad1 = (bf16t*)alloc((size_t)BATCH*PH*PH*CIN*2);   // padded NHWC input, bf16
  bf16t* wRt   = (bf16t*)alloc((size_t)K_RED*CH*2);
  bf16t* wKt   = (bf16t*)alloc((size_t)K_CH*CH*2);
  bf16t* wVt   = (bf16t*)alloc((size_t)K_CH*CH*2);
  bf16t* wOt   = (bf16t*)alloc((size_t)K_CH*CH*2);
  float* xf    = (float*)alloc((size_t)BATCH*CH*HWXY*4);     // x (reduce conv), f32 NCHW
  bf16t* xpad2 = (bf16t*)alloc((size_t)BATCH*PH*PH*CH*2);    // x padded NHWC bf16
  bf16t* qb    = (bf16t*)alloc((size_t)HWXY*BATCH*CH*2);
  bf16t* kb    = (bf16t*)alloc((size_t)HWXY*BATCH*CH*2);
  bf16t* vt    = (bf16t*)alloc((size_t)HWXY*CH*BATCH*2);
  float* att   = (float*)alloc((size_t)HWXY*BATCH*BATCH*4);
  bf16t* attb  = (bf16t*)alloc((size_t)HWXY*BATCH*BATCH*2);
  float* virt  = (float*)alloc((size_t)BATCH*HWXY*CH*4);
  bf16t* vpad  = (bf16t*)alloc((size_t)BATCH*PH*PH*CH*2);

  auto zgrid = [](long n16) { return (unsigned)((n16 + 255) / 256); };
  long n16_1 = (long)BATCH*PH*PH*CIN*2/16;
  long n16_2 = (long)BATCH*PH*PH*CH*2/16;
  zero_b16<<<zgrid(n16_1), 256, 0, stream>>>((uint4*)xpad1, n16_1);
  zero_b16<<<zgrid(n16_2), 256, 0, stream>>>((uint4*)xpad2, n16_2);
  zero_b16<<<zgrid(n16_2), 256, 0, stream>>>((uint4*)vpad,  n16_2);

  fill_xpad1<<<(unsigned)(((long)BATCH*CIN*HWXY + 255)/256), 256, 0, stream>>>(bbox, xpad1);
  transpose_w<<<(unsigned)(((long)CH*CIN*9 + 255)/256), 256, 0, stream>>>(w_red, wRt, CIN);
  transpose_w<<<(unsigned)(((long)CH*CH*9 + 255)/256), 256, 0, stream>>>(w_k, wKt, CH);
  transpose_w<<<(unsigned)(((long)CH*CH*9 + 255)/256), 256, 0, stream>>>(w_v, wVt, CH);
  transpose_w<<<(unsigned)(((long)CH*CH*9 + 255)/256), 256, 0, stream>>>(w_out, wOt, CH);
  compute_q<<<(unsigned)(((long)BATCH*HWXY*CH + 255)/256), 256, 0, stream>>>(status, rois, w_q, qb);

  dim3 blk(256);
  { // reduce conv: x = conv(bbox, w_reduce)
    GemmP p{}; p.A = xpad1; p.Bt = wRt; p.outF = xf; p.outB = xpad2; p.resid = nullptr;
    p.M = M_CONV; p.N = CH; p.K = K_RED; p.zA = 0; p.zB = 0; p.scale = 1.f;
    gemm_bf16_wmma<0><<<dim3(M_CONV/128, CH/128, 1), blk, 0, stream>>>(p);
  }
  { // k conv
    GemmP p{}; p.A = xpad2; p.Bt = wKt; p.outF = nullptr; p.outB = kb; p.resid = nullptr;
    p.M = M_CONV; p.N = CH; p.K = K_CH; p.zA = 0; p.zB = 0; p.scale = 1.f;
    gemm_bf16_wmma<1><<<dim3(M_CONV/128, CH/128, 1), blk, 0, stream>>>(p);
  }
  { // v conv
    GemmP p{}; p.A = xpad2; p.Bt = wVt; p.outF = nullptr; p.outB = vt; p.resid = nullptr;
    p.M = M_CONV; p.N = CH; p.K = K_CH; p.zA = 0; p.zB = 0; p.scale = 1.f;
    gemm_bf16_wmma<2><<<dim3(M_CONV/128, CH/128, 1), blk, 0, stream>>>(p);
  }
  { // att[b][j] per hw = q . k / sqrt(512)
    GemmP p{}; p.A = qb; p.Bt = kb; p.outF = att; p.outB = nullptr; p.resid = nullptr;
    p.M = BATCH; p.N = BATCH; p.K = CH;
    p.zA = (long)BATCH*CH; p.zB = (long)BATCH*CH; p.scale = 0.04419417382415922f; // 1/sqrt(512)
    gemm_bf16_wmma<3><<<dim3(BATCH/128, BATCH/128, HWXY), blk, 0, stream>>>(p);
  }
  softmax_rows<<<HWXY*BATCH, 256, 0, stream>>>(att, attb);
  { // virt[b][c] per hw = att . v
    GemmP p{}; p.A = attb; p.Bt = vt; p.outF = virt; p.outB = nullptr; p.resid = nullptr;
    p.M = BATCH; p.N = CH; p.K = BATCH;
    p.zA = (long)BATCH*BATCH; p.zB = (long)CH*BATCH; p.scale = 1.f;
    gemm_bf16_wmma<4><<<dim3(BATCH/128, CH/128, HWXY), blk, 0, stream>>>(p);
  }
  layernorm_relu_pad<<<BATCH, 256, 0, stream>>>(virt, gamma, beta, vpad);
  { // out conv + residual
    GemmP p{}; p.A = vpad; p.Bt = wOt; p.outF = out; p.outB = nullptr; p.resid = xf;
    p.M = M_CONV; p.N = CH; p.K = K_CH; p.zA = 0; p.zB = 0; p.scale = 1.f;
    gemm_bf16_wmma<5><<<dim3(M_CONV/128, CH/128, 1), blk, 0, stream>>>(p);
  }
}
